// FuncSelfAttention_72464688218819
// MI455X (gfx1250) — matrix operand order repaired and checked by reading, not compiled, not run on hardware
//
#include <hip/hip_runtime.h>

typedef __attribute__((ext_vector_type(16))) _Float16 v16h;
typedef __attribute__((ext_vector_type(8)))  _Float16 v8h;
typedef __attribute__((ext_vector_type(8)))  float    v8f;

#define WMMA_F32_F16(a,b,c) \
  __builtin_amdgcn_wmma_f32_16x16x32_f16(false,(a),false,(b),(short)0,(c),false,false)

// A-matrix (16x32 f16): lane ln holds row M=ln; slots 0..7 = K {hi*8 .. hi*8+7},
// slots 8..15 = K {16+hi*8 .. 16+hi*8+7}  (per ISA 7.12.2 16-bit A layout)
__device__ __forceinline__ v16h load_a_f16(const _Float16* rowk, int hi) {
  v8h lo = *(const v8h*)(rowk + hi * 8);
  v8h hh = *(const v8h*)(rowk + 16 + hi * 8);
  v16h r;
#pragma unroll
  for (int i = 0; i < 8; ++i) { r[i] = lo[i]; r[i + 8] = hh[i]; }
  return r;
}
// B-matrix (32x16 f16): lane ln holds col N=ln; slots 0..7 = K {hi*8 .. hi*8+7},
// slots 8..15 = K {16+hi*8 .. 16+hi*8+7}
__device__ __forceinline__ v16h load_b_f16(const _Float16* rowk, int hi) {
  v8h lo = *(const v8h*)(rowk + hi * 8);
  v8h hh = *(const v8h*)(rowk + 16 + hi * 8);
  v16h r;
#pragma unroll
  for (int i = 0; i < 8; ++i) { r[i] = lo[i]; r[i + 8] = hh[i]; }
  return r;
}

// Async global->LDS copy, 16B per lane. LDS operand is the 32-bit LDS byte
// offset (flat LDS addresses carry the offset in addr[31:0], ISA 10.2).
// Tracked with ASYNCcnt; fence with s_wait_asynccnt before consuming.
__device__ __forceinline__ void async_ld_b128(void* lds_dst, const void* gsrc) {
  unsigned int loff = (unsigned int)(unsigned long long)lds_dst;
  asm volatile("global_load_async_to_lds_b128 %0, %1, off"
               :: "v"(loff), "v"(gsrc) : "memory");
}
__device__ __forceinline__ void async_wait0() {
  asm volatile("s_wait_asynccnt 0x0" ::: "memory");
}

// ---------------------------------------------------------------------------
// Kernel 1: QKV projection (per (b,s): [1024 pix x 64 c] @ [64 c x 192 o])
// Writes Q,K as [b, head, s, d] fp16 and V transposed as [b, head, d, s] fp16.
// head = (sx*2+sy)*4 + ch ;  d = dc*256 + xi*16 + yi
// ---------------------------------------------------------------------------
__global__ void k_qkv(const float* __restrict__ seq,
                      const float* __restrict__ w_qkv,
                      _Float16* __restrict__ Q,
                      _Float16* __restrict__ Kb,
                      _Float16* __restrict__ Vt) {
  extern __shared__ char smem[];
  _Float16* Xs = (_Float16*)smem;                  // [1024 pix][64 c]
  _Float16* Ws = (_Float16*)(smem + 1024 * 64 * 2);// [192 o][64 c]
  const int tid = threadIdx.x;
  const int b = blockIdx.x >> 8;
  const int s = blockIdx.x & 255;

  const float4* src4 = (const float4*)(seq + (size_t)blockIdx.x * 65536);
  for (int i = tid; i < 16384; i += 256) {           // transpose [c][pix] -> [pix][c]
    float4 f = src4[i];
    int c = i >> 8;
    int pix = (i & 255) << 2;
    Xs[(pix + 0) * 64 + c] = (_Float16)f.x;
    Xs[(pix + 1) * 64 + c] = (_Float16)f.y;
    Xs[(pix + 2) * 64 + c] = (_Float16)f.z;
    Xs[(pix + 3) * 64 + c] = (_Float16)f.w;
  }
  for (int i = tid; i < 192 * 64; i += 256) Ws[i] = (_Float16)w_qkv[i];
  __syncthreads();

  const int wave = tid >> 5, lane = tid & 31, ln = lane & 15, hi = lane >> 4;
  for (int j = wave; j < 768; j += 8) {              // 64 pix-tiles x 12 o-tiles
    int ot = j % 12, pt = j / 12;
    v8f acc = {0.f, 0.f, 0.f, 0.f, 0.f, 0.f, 0.f, 0.f};
#pragma unroll
    for (int kk = 0; kk < 64; kk += 32) {
      v16h a  = load_a_f16(&Xs[(pt * 16 + ln) * 64 + kk], hi); // M=pix, K=c
      v16h bm = load_b_f16(&Ws[(ot * 16 + ln) * 64 + kk], hi); // K=c,  N=o
      acc = WMMA_F32_F16(a, bm, acc);
    }
    // D lane: n = o (fixed), m = 8 consecutive pixels (hi*8 + v)
    int o = ot * 16 + ln;
    int split = o >> 6, oc = o & 63, ch = oc >> 4, dc = oc & 15;
    int pix0 = pt * 16 + hi * 8;
    int h = pix0 >> 5, wc = pix0 & 31;
    int sx = h >> 4, xi = h & 15, sy = wc >> 4, yi0 = wc & 15;
    int head = (sx * 2 + sy) * 4 + ch;
    int d0 = dc * 256 + xi * 16 + yi0;
    size_t bh = (size_t)(b * 16 + head);
    if (split < 2) {
      _Float16* dst = (split == 0 ? Q : Kb) + (bh * 256 + s) * 4096 + d0;
      v8h hv;
#pragma unroll
      for (int v = 0; v < 8; ++v) hv[v] = (_Float16)acc[v];
      *(v8h*)dst = hv;                               // 8 consecutive d
    } else {
#pragma unroll
      for (int v = 0; v < 8; ++v)
        Vt[(bh * 4096 + (size_t)(d0 + v)) * 256 + s] = (_Float16)acc[v];
    }
  }
}

// ---------------------------------------------------------------------------
// Kernel 2: attention per (b,head, 64-query tile). D = 4096, keys = 256.
// Phase 1: scores[64,256] = Q Kt (stream d in 128-chunks via async->LDS),
//          f32 softmax.
// Phase 2: sa chunk D[d,q] = Vt[d,keys] x attnT[keys,q] -> contiguous d stores
// ---------------------------------------------------------------------------
__global__ void k_attn(const _Float16* __restrict__ Q,
                       const _Float16* __restrict__ Kg_,
                       const _Float16* __restrict__ Vt,
                       _Float16* __restrict__ SA) {
  extern __shared__ char smem[];
  _Float16* Qc = (_Float16*)smem;               // [64][128]   (bytes 0..16K)
  _Float16* Kc = (_Float16*)(smem + 16384);     // [256][128]  (16K..80K)
  float*    Sc = (float*)smem;                  // [64][256]   (reuse 0..64K)
  _Float16* At = (_Float16*)(smem + 65536);     // [64][256]   (64K..96K)
  _Float16* Vc = (_Float16*)smem;               // [128][256]  (reuse 0..64K)

  const int tid = threadIdx.x;
  const int wave = tid >> 5, lane = tid & 31, ln = lane & 15, hi = lane >> 4;
  const int bh = blockIdx.x >> 2;
  const int q0 = (blockIdx.x & 3) * 64;
  const _Float16* Qg = Q   + (size_t)bh * 256 * 4096 + (size_t)q0 * 4096;
  const _Float16* Kg = Kg_ + (size_t)bh * 256 * 4096;
  const _Float16* Vg = Vt  + (size_t)bh * 4096 * 256;
  _Float16*       Og = SA  + (size_t)bh * 256 * 4096 + (size_t)q0 * 4096;

  const int mT = wave >> 1, ntBase = (wave & 1) * 8;   // wave owns 1 q-tile x 8 key-tiles
  v8f z = {0.f, 0.f, 0.f, 0.f, 0.f, 0.f, 0.f, 0.f};
  v8f acc[8];
#pragma unroll
  for (int t = 0; t < 8; ++t) acc[t] = z;

  for (int chunk = 0; chunk < 32; ++chunk) {           // stream over d
    const int d0 = chunk * 128;
    for (int i = tid; i < 1024; i += 256) {            // uniform trip: EXEC all-1s
      int r = i >> 4, cv = (i & 15) * 8;
      async_ld_b128(&Qc[r * 128 + cv], &Qg[(size_t)r * 4096 + d0 + cv]);
    }
    for (int i = tid; i < 4096; i += 256) {
      int r = i >> 4, cv = (i & 15) * 8;
      async_ld_b128(&Kc[r * 128 + cv], &Kg[(size_t)r * 4096 + d0 + cv]);
    }
    async_wait0();
    __syncthreads();
#pragma unroll
    for (int kk = 0; kk < 4; ++kk) {
      v16h a = load_a_f16(&Qc[(mT * 16 + ln) * 128 + kk * 32], hi); // M=q, K=d
#pragma unroll
      for (int t = 0; t < 8; ++t) {
        v16h bm = load_b_f16(&Kc[((ntBase + t) * 16 + ln) * 128 + kk * 32], hi); // K=d, N=key
        acc[t] = WMMA_F32_F16(a, bm, acc[t]);
      }
    }
    __syncthreads();
  }

  const float scale = 1.0f / 64.0f;                    // 1/sqrt(16*16*16)
#pragma unroll
  for (int t = 0; t < 8; ++t)
#pragma unroll
    for (int v = 0; v < 8; ++v)
      Sc[(mT * 16 + hi * 8 + v) * 256 + (ntBase + t) * 16 + ln] = acc[t][v] * scale;
  __syncthreads();

  if (tid < 64) {                                      // row softmax
    float mx = -1e30f;
    for (int j = 0; j < 256; ++j) mx = fmaxf(mx, Sc[tid * 256 + j]);
    float sum = 0.f;
    for (int j = 0; j < 256; ++j) {
      float e = __expf(Sc[tid * 256 + j] - mx);
      Sc[tid * 256 + j] = e;
      sum += e;
    }
    float inv = 1.0f / sum;
    for (int j = 0; j < 256; ++j)
      At[tid * 256 + j] = (_Float16)(Sc[tid * 256 + j] * inv);
  }
  __syncthreads();

  for (int chunk = 0; chunk < 32; ++chunk) {           // sa = attn @ V, chunked over d
    const int d0 = chunk * 128;
    for (int i = tid; i < 4096; i += 256) {
      int r = i >> 5, cv = (i & 31) * 8;
      async_ld_b128(&Vc[r * 256 + cv], &Vg[(size_t)(d0 + r) * 256 + cv]);
    }
    async_wait0();
    __syncthreads();
    v8f acc2[4];
#pragma unroll
    for (int t = 0; t < 4; ++t) acc2[t] = z;
#pragma unroll
    for (int kk = 0; kk < 8; ++kk) {
      v16h a = load_a_f16(&Vc[(wave * 16 + ln) * 256 + kk * 32], hi);      // M=d, K=key
#pragma unroll
      for (int qt = 0; qt < 4; ++qt) {
        v16h bm = load_b_f16(&At[(qt * 16 + ln) * 256 + kk * 32], hi);     // K=key, N=q
        acc2[qt] = WMMA_F32_F16(a, bm, acc2[qt]);
      }
    }
#pragma unroll
    for (int qt = 0; qt < 4; ++qt) {                   // lane: fixed q, 8 consecutive d
      v8h hv;
#pragma unroll
      for (int v = 0; v < 8; ++v) hv[v] = (_Float16)acc2[qt][v];
      *(v8h*)&Og[(size_t)(qt * 16 + ln) * 4096 + d0 + wave * 16 + hi * 8] = hv;
    }
    __syncthreads();
  }
}

// ---------------------------------------------------------------------------
// Kernel 3: output projection: out[b,s,o,h,w] = w_out[o,:] . sa[b,s,:,h,w] + b[o]
// Gathers SA[b,head,s,d] back to [pix][c] in LDS, then WMMA.
// ---------------------------------------------------------------------------
__global__ void k_out(const _Float16* __restrict__ SA,
                      const float* __restrict__ w_out,
                      const float* __restrict__ b_out,
                      float* __restrict__ out) {
  extern __shared__ char smem[];
  _Float16* Xs = (_Float16*)smem;                   // [1024 pix][64 c]
  _Float16* Wo = (_Float16*)(smem + 1024 * 64 * 2); // [64 o][64 c]
  const int tid = threadIdx.x;
  const int b = blockIdx.x >> 8;
  const int s = blockIdx.x & 255;

  for (int i = tid; i < 4096; i += 256) Wo[i] = (_Float16)w_out[i];
  for (int i = tid; i < 8192; i += 256) {
    int head = i >> 9;
    int dv = (i & 511) * 8;
    v8h hv = *(const v8h*)&SA[(((size_t)(b * 16 + head)) * 256 + s) * 4096 + dv];
    int ch = head & 3, sh = head >> 2, sx = sh >> 1, sy = sh & 1;
    int dc = dv >> 8, rem = dv & 255, xi = rem >> 4, yi0 = rem & 15;
    int c = ch * 16 + dc;
    int pix0 = (sx * 16 + xi) * 32 + sy * 16 + yi0;
#pragma unroll
    for (int v = 0; v < 8; ++v) Xs[(pix0 + v) * 64 + c] = hv[v];
  }
  __syncthreads();

  const int wave = tid >> 5, lane = tid & 31, ln = lane & 15, hi = lane >> 4;
  for (int j = wave; j < 256; j += 8) {             // 64 pix-tiles x 4 o-tiles
    int ot = j & 3, pt = j >> 2;
    v8f acc = {0.f, 0.f, 0.f, 0.f, 0.f, 0.f, 0.f, 0.f};
#pragma unroll
    for (int kk = 0; kk < 64; kk += 32) {
      v16h a  = load_a_f16(&Xs[(pt * 16 + ln) * 64 + kk], hi);
      v16h bm = load_b_f16(&Wo[(ot * 16 + ln) * 64 + kk], hi);
      acc = WMMA_F32_F16(a, bm, acc);
    }
    int o = ot * 16 + ln;
    float bias = b_out[o];
    int pix0 = pt * 16 + hi * 8;
    float* dst = out + (((size_t)(b * 256 + s)) * 64 + o) * 1024 + pix0;
    float4 f0 = make_float4(acc[0] + bias, acc[1] + bias, acc[2] + bias, acc[3] + bias);
    float4 f1 = make_float4(acc[4] + bias, acc[5] + bias, acc[6] + bias, acc[7] + bias);
    *(float4*)(dst + 0) = f0;
    *(float4*)(dst + 4) = f1;
  }
}

extern "C" void kernel_launch(void* const* d_in, const int* in_sizes, int n_in,
                              void* d_out, int out_size, void* d_ws, size_t ws_size,
                              hipStream_t stream) {
  (void)in_sizes; (void)n_in; (void)out_size; (void)ws_size;
  const float* seq   = (const float*)d_in[0];
  const float* w_qkv = (const float*)d_in[1];
  const float* w_out = (const float*)d_in[2];
  const float* b_out = (const float*)d_in[3];
  char* ws = (char*)d_ws;
  const size_t SEG = (size_t)64 << 20;              // 64 MiB per buffer
  _Float16* Q  = (_Float16*)(ws);
  _Float16* Kb = (_Float16*)(ws + SEG);
  _Float16* Vt = (_Float16*)(ws + 2 * SEG);
  _Float16* SA = (_Float16*)(ws + 3 * SEG);
  float* out = (float*)d_out;

  k_qkv <<<512, 256, (1024 * 64 + 192 * 64) * 2, stream>>>(seq, w_qkv, Q, Kb, Vt);
  k_attn<<<128, 256, 96 * 1024,                  stream>>>(Q, Kb, Vt, SA);
  k_out <<<512, 256, (1024 * 64 + 64 * 64) * 2,  stream>>>(SA, w_out, b_out, out);
}
